// Transformer_10118942949799
// MI455X (gfx1250) — matrix operand order repaired and checked by reading, not compiled, not run on hardware
//
#include <hip/hip_runtime.h>
#include <math.h>

typedef __attribute__((ext_vector_type(16))) _Float16 v16h;
typedef __attribute__((ext_vector_type(8)))  float    v8f;

#define INV_SQRT3f 0.57735026918962576f

// branch-free fast tanh: 1 - 2/(exp(2x)+1); saturates to +-1 without branches
__device__ __forceinline__ float fast_tanh(float x) {
    float t = __builtin_amdgcn_exp2f(x * 2.885390081777927f); // exp(2x)
    return 1.0f - 2.0f * __builtin_amdgcn_rcpf(t + 1.0f);
}
__device__ __forceinline__ float gelu_t(float x) {
    const float c0 = 0.7978845608028654f; // sqrt(2/pi)
    return 0.5f * x * (1.0f + fast_tanh(c0 * (x + 0.044715f * x * x * x)));
}
__device__ __forceinline__ float fast_exp(float x) {
    return __builtin_amdgcn_exp2f(x * 1.4426950408889634f);
}

__device__ __forceinline__ v8f wmma16(v16h a, v16h b, v8f c) {
    // D = A(16x32 f16) x B(32x16 f16) + C(16x16 f32)
    return __builtin_amdgcn_wmma_f32_16x16x32_f16(false, a, false, b, (short)0, c,
                                                  false, false);
}

// A-matrix (16x32 f16) K index for vgpr half-pair j, lane-half hi
__device__ __forceinline__ int akidx(int j, int hi) {
    return 2 * j + (j < 4 ? 0 : 8) + (hi ? 8 : 0);
}

// Packed B chunk: 32 contiguous bytes per lane -> two global_load_b128
__device__ __forceinline__ v16h loadBpk(const _Float16* pk, int C, int kc, int c) {
    return ((const v16h*)pk)[((size_t)kc * C + c) * 32 + threadIdx.x];
}

// A chunk (K=32 slice at kofs) from LDS row-major [16][64] f16
__device__ __forceinline__ v16h loadA_lds(const _Float16* h, int mrow, int kofs, int hi) {
    v16h a;
#pragma unroll
    for (int j = 0; j < 8; ++j) {
        int k = kofs + akidx(j, hi);
        a[2 * j]     = h[mrow * 64 + k];
        a[2 * j + 1] = h[mrow * 64 + k + 1];
    }
    return a;
}

// ---------------------------------------------------------------------------
// KP: pack a row-major W[K x ld] (f32) into WMMA-B f16 layout.
// out[((kc*C + c)*32 + l)*16 + s] = W[(kc*32 + (l>>4)*8 + (s&7) + (s>>3)*16)*ld + c*16 + (l&15)]
__global__ void kp_pack(const float* __restrict__ W, _Float16* __restrict__ out,
                        int K, int ld) {
    int idx = blockIdx.x * blockDim.x + threadIdx.x;
    if (idx >= K * ld) return;
    int s = idx & 15;
    int l = (idx >> 4) & 31;
    int chunk = idx >> 9;
    int C = ld >> 4;
    int kc = chunk / C, c = chunk % C;
    int k = kc * 32 + (l >> 4) * 8 + (s & 7) + (s >> 3) * 16;
    int col = c * 16 + (l & 15);
    out[idx] = (_Float16)W[(size_t)k * ld + col];
}

// KP: pack path weights Wp(4,64,32), logical col = p*32+u (K=64, 128 cols)
__global__ void kp_pack_path(const float* __restrict__ Wp, _Float16* __restrict__ out) {
    int idx = blockIdx.x * blockDim.x + threadIdx.x;
    if (idx >= 8192) return;
    int s = idx & 15;
    int l = (idx >> 4) & 31;
    int chunk = idx >> 9;           // kc*8 + c
    int kc = chunk >> 3, c = chunk & 7;
    int k = kc * 32 + (l >> 4) * 8 + (s & 7) + (s >> 3) * 16;
    int col = c * 16 + (l & 15);
    int p = col >> 5, u = col & 31;
    out[idx] = (_Float16)Wp[(size_t)p * 2048 + k * 32 + u];
}

// ---------------------------------------------------------------------------
// K0: zero fill
__global__ void k0_zero(float* __restrict__ p, size_t n) {
    size_t i = (size_t)blockIdx.x * blockDim.x + threadIdx.x;
    if (i < n) p[i] = 0.0f;
}

// ---------------------------------------------------------------------------
// K1: per 16-edge tile (one wave32): MLP_k, MLP_v (WMMA), uvu TP for k,
// logit GEMMs (WMMA) + scatter-max. wp_v spilled to workspace for K3.
__global__ __launch_bounds__(32) void k1_edge_wmma(
    const float* __restrict__ node_s, const float* __restrict__ node_v,
    const float* __restrict__ edge_attr, const float* __restrict__ esa,
    const _Float16* __restrict__ pkWk0, const _Float16* __restrict__ pkWk1,
    const _Float16* __restrict__ pkWkp, const _Float16* __restrict__ pkWv0,
    const _Float16* __restrict__ pkWv1, const _Float16* __restrict__ pkWvp,
    const _Float16* __restrict__ pkWls, const _Float16* __restrict__ pkWlv,
    const int* __restrict__ edge_src, const int* __restrict__ edge_dst,
    float* __restrict__ wpv_out, float* __restrict__ logit_out,
    unsigned int* __restrict__ lmax, int E)
{
    __shared__ _Float16 hstage[16 * 64];   // GEMM relayout staging (f16)
    __shared__ float    wpk[16 * 128];     // k-side path-projected weights
    __shared__ float    ksL[16 * 64];      // k_s tile
    __shared__ float    kvL[16 * 192];     // k_v tile (v,d)
    __shared__ float    mst[16 * 128];     // m_s / m_v staging

    const int l  = threadIdx.x;
    const int nl = l & 15;
    const int hi = l >> 4;
    const int e0 = blockIdx.x * 16;

    // ---- X tile (edge_scalar_attr 16x32) in A layout
    v16h ax;
    {
        int er = e0 + nl; if (er >= E) er = E - 1;
        const float* xrow = esa + (size_t)er * 32;
#pragma unroll
        for (int j = 0; j < 8; ++j) {
            int k = akidx(j, hi);
            ax[2 * j]     = (_Float16)xrow[k];
            ax[2 * j + 1] = (_Float16)xrow[k + 1];
        }
    }

    // ---- both MLP sides: 0 = k (keep wp in LDS), 1 = v (spill wp to global)
    for (int side = 0; side < 2; ++side) {
        const _Float16* W0 = side ? pkWv0 : pkWk0;
        const _Float16* W1 = side ? pkWv1 : pkWk1;
        const _Float16* Wp = side ? pkWvp : pkWkp;

        // H0 = gelu(X @ W0 / sqrt(32))   (16x64)
#pragma unroll
        for (int c = 0; c < 4; ++c) {
            v16h b = loadBpk(W0, 4, 0, c);
            v8f acc = {};
            acc = wmma16(ax, b, acc);
#pragma unroll
            for (int i = 0; i < 8; ++i)
                hstage[(i + 8 * hi) * 64 + c * 16 + nl] =
                    (_Float16)gelu_t(acc[i] * 0.1767766952966369f);
        }
        __syncthreads();
        v16h a0 = loadA_lds(hstage, nl, 0, hi);
        v16h a1 = loadA_lds(hstage, nl, 32, hi);
        __syncthreads();

        // H1 = gelu(H0 @ W1 / 8)   (16x64)
#pragma unroll
        for (int c = 0; c < 4; ++c) {
            v16h b0 = loadBpk(W1, 4, 0, c);
            v16h b1 = loadBpk(W1, 4, 1, c);
            v8f acc = {};
            acc = wmma16(a0, b0, acc);
            acc = wmma16(a1, b1, acc);
#pragma unroll
            for (int i = 0; i < 8; ++i)
                hstage[(i + 8 * hi) * 64 + c * 16 + nl] =
                    (_Float16)gelu_t(acc[i] * 0.125f);
        }
        __syncthreads();
        a0 = loadA_lds(hstage, nl, 0, hi);
        a1 = loadA_lds(hstage, nl, 32, hi);
        __syncthreads();

        // WP = H1 @ Wpath / 8   (16x128, cols = p*32+u)
#pragma unroll
        for (int c = 0; c < 8; ++c) {
            v16h b0 = loadBpk(Wp, 8, 0, c);
            v16h b1 = loadBpk(Wp, 8, 1, c);
            v8f acc = {};
            acc = wmma16(a0, b0, acc);
            acc = wmma16(a1, b1, acc);
            const int col = c * 16 + nl;
            if (side == 0) {
#pragma unroll
                for (int i = 0; i < 8; ++i)
                    wpk[(i + 8 * hi) * 128 + col] = acc[i] * 0.125f;
            } else {
#pragma unroll
                for (int i = 0; i < 8; ++i) {
                    int er = e0 + i + 8 * hi;
                    if (er < E) wpv_out[(size_t)er * 128 + col] = acc[i] * 0.125f;
                }
            }
        }
        __syncthreads();
    }

    // ---- uvu tensor product for k side: 2 lanes per edge (hi splits v-range)
    const int   eR   = (e0 + nl < E) ? (e0 + nl) : (E - 1);
    const int   srcn = edge_src[eR];
    const int   dstn = edge_dst[eR];
    const float shs  = edge_attr[(size_t)eR * 4 + 0];
    const float shv0 = edge_attr[(size_t)eR * 4 + 1];
    const float shv1 = edge_attr[(size_t)eR * 4 + 2];
    const float shv2 = edge_attr[(size_t)eR * 4 + 3];

    if (hi == 0) {
#pragma unroll 4
        for (int u = 0; u < 32; ++u) {
            float ss = node_s[(size_t)srcn * 32 + u];
            float w0 = wpk[nl * 128 + u];        // path0 -> s0
            float w1 = wpk[nl * 128 + 32 + u];   // path1 -> v0
            ksL[nl * 64 + u] = w0 * ss * shs;
            kvL[nl * 192 + u * 3 + 0] = w1 * ss * shv0;
            kvL[nl * 192 + u * 3 + 1] = w1 * ss * shv1;
            kvL[nl * 192 + u * 3 + 2] = w1 * ss * shv2;
        }
    } else {
#pragma unroll 4
        for (int u = 0; u < 32; ++u) {
            const float* vp = node_v + ((size_t)srcn * 32 + u) * 3;
            float x0 = vp[0], x1 = vp[1], x2 = vp[2];
            float w2 = wpk[nl * 128 + 64 + u];   // path2 -> v1
            float w3 = wpk[nl * 128 + 96 + u];   // path3 -> s1
            ksL[nl * 64 + 32 + u] = w3 * (x0 * shv0 + x1 * shv1 + x2 * shv2) * INV_SQRT3f;
            kvL[nl * 192 + (32 + u) * 3 + 0] = w2 * x0 * shs;
            kvL[nl * 192 + (32 + u) * 3 + 1] = w2 * x1 * shs;
            kvL[nl * 192 + (32 + u) * 3 + 2] = w2 * x2 * shs;
        }
    }
    __syncthreads();

    // ---- logits:  m_s = dst_s @ Wl_s(32x128 view), logit_s = k_s . m_s[:, v*2+h]
    float logit = 0.0f;
    v16h adst;
#pragma unroll
    for (int j = 0; j < 8; ++j) {
        int k = akidx(j, hi);
        adst[2 * j]     = (_Float16)node_s[(size_t)dstn * 32 + k];
        adst[2 * j + 1] = (_Float16)node_s[(size_t)dstn * 32 + k + 1];
    }
#pragma unroll
    for (int c = 0; c < 8; ++c) {
        v16h b = loadBpk(pkWls, 8, 0, c);
        v8f acc = {};
        acc = wmma16(adst, b, acc);
#pragma unroll
        for (int i = 0; i < 8; ++i)
            mst[(i + 8 * hi) * 128 + c * 16 + nl] = acc[i];
    }
    __syncthreads();
#pragma unroll 8
    for (int v = 0; v < 64; ++v)
        logit += ksL[nl * 64 + v] * mst[nl * 128 + v * 2 + hi];

    float logitv = 0.0f;
    for (int d = 0; d < 3; ++d) {
        __syncthreads();
        v16h adv;
#pragma unroll
        for (int j = 0; j < 8; ++j) {
            int k = akidx(j, hi);
            adv[2 * j]     = (_Float16)node_v[((size_t)dstn * 32 + k) * 3 + d];
            adv[2 * j + 1] = (_Float16)node_v[((size_t)dstn * 32 + k + 1) * 3 + d];
        }
#pragma unroll
        for (int c = 0; c < 8; ++c) {
            v16h b = loadBpk(pkWlv, 8, 0, c);
            v8f acc = {};
            acc = wmma16(adv, b, acc);
#pragma unroll
            for (int i = 0; i < 8; ++i)
                mst[(i + 8 * hi) * 128 + c * 16 + nl] = acc[i];
        }
        __syncthreads();
#pragma unroll 8
        for (int v = 0; v < 64; ++v)
            logitv += kvL[nl * 192 + v * 3 + d] * mst[nl * 128 + v * 2 + hi];
    }
    logit += logitv * INV_SQRT3f;

    if (e0 + nl < E) {
        logit_out[(size_t)(e0 + nl) * 2 + hi] = logit;
        // baseline of scatter-max is 0.0 (bits 0); positive f32 order == uint order
        if (logit > 0.0f)
            atomicMax(lmax + (size_t)dstn * 2 + hi, __float_as_uint(logit));
    }
}

// ---------------------------------------------------------------------------
// K2: ex = cutoff * exp(logit - lmax[dst]); z[dst] += ex
__global__ void k2_softmax_den(const float* __restrict__ logit,
                               const unsigned int* __restrict__ lmax,
                               const float* __restrict__ cutoff,
                               const int* __restrict__ edge_dst,
                               float* __restrict__ ex, float* __restrict__ z, int E)
{
    int t = blockIdx.x * blockDim.x + threadIdx.x;
    if (t >= E * 2) return;
    int e = t >> 1, h = t & 1;
    int d = edge_dst[e];
    float lm = __uint_as_float(lmax[(size_t)d * 2 + h]);
    float v  = cutoff[e] * fast_exp(logit[t] - lm);
    ex[t] = v;
    atomicAdd(z + (size_t)d * 2 + h, v);
}

// ---------------------------------------------------------------------------
// K3: one lane per (edge, v-channel): recompute v-side uvu TP from wp_v,
// scale by sqrt(relu(alpha)), scatter-add to node accumulators.
__global__ void k3_scatter(const float* __restrict__ wpv, const float* __restrict__ ex,
                           const float* __restrict__ z,
                           const float* __restrict__ node_s, const float* __restrict__ node_v,
                           const float* __restrict__ edge_attr,
                           const int* __restrict__ edge_src, const int* __restrict__ edge_dst,
                           float* __restrict__ out_s, float* __restrict__ out_v, int E)
{
    int t = blockIdx.x * blockDim.x + threadIdx.x;
    if (t >= E * 64) return;
    const int e = t >> 6, v = t & 63;
    const int h = v >> 5;
    const int dn = edge_dst[e];
    float zz = z[(size_t)dn * 2 + h];
    if (zz == 0.0f) zz = 1.0f;
    const float alpha = ex[(size_t)e * 2 + h] * __builtin_amdgcn_rcpf(zz);
    const float a = sqrtf(fmaxf(alpha, 0.0f));

    const int sn = edge_src[e];
    const float shs  = edge_attr[(size_t)e * 4 + 0];
    const float shv0 = edge_attr[(size_t)e * 4 + 1];
    const float shv1 = edge_attr[(size_t)e * 4 + 2];
    const float shv2 = edge_attr[(size_t)e * 4 + 3];
    const float* wp  = wpv + (size_t)e * 128;

    float vs, vv0, vv1, vv2;
    if (v < 32) {
        float ss = node_s[(size_t)sn * 32 + v];
        vs = wp[v] * ss * shs;                       // path0
        float w1 = wp[32 + v];                       // path1
        vv0 = w1 * ss * shv0; vv1 = w1 * ss * shv1; vv2 = w1 * ss * shv2;
    } else {
        int u = v - 32;
        const float* xp = node_v + ((size_t)sn * 32 + u) * 3;
        float x0 = xp[0], x1 = xp[1], x2 = xp[2];
        vs = wp[96 + u] * (x0 * shv0 + x1 * shv1 + x2 * shv2) * INV_SQRT3f; // path3
        float w2 = wp[64 + u];                       // path2
        vv0 = w2 * x0 * shs; vv1 = w2 * x1 * shs; vv2 = w2 * x2 * shs;
    }
    vs *= a; vv0 *= a; vv1 *= a; vv2 *= a;
    atomicAdd(out_s + (size_t)dn * 64 + v, vs);
    atomicAdd(out_v + (size_t)dn * 192 + v * 3 + 0, vv0);
    atomicAdd(out_v + (size_t)dn * 192 + v * 3 + 1, vv1);
    atomicAdd(out_v + (size_t)dn * 192 + v * 3 + 2, vv2);
}

// ---------------------------------------------------------------------------
// K4: per 16-node tile (one wave32): ys = out_s@Wo_s/8, yv = out_v[:,:,d]@Wo_v/8
__global__ __launch_bounds__(32) void k4_node_wmma(
    const float* __restrict__ out_s, const float* __restrict__ out_v,
    const _Float16* __restrict__ pkWos, const _Float16* __restrict__ pkWov,
    float* __restrict__ out, int N)
{
    const int l = threadIdx.x, nl = l & 15, hi = l >> 4;
    const int n0 = blockIdx.x * 16;
    int rR = n0 + nl; if (rR >= N) rR = N - 1;
    const float sc = 0.125f; // 1/sqrt(64)

    v16h a0, a1;
    {
        const float* s = out_s + (size_t)rR * 64;
#pragma unroll
        for (int j = 0; j < 8; ++j) {
            int k = akidx(j, hi);
            a0[2 * j] = (_Float16)s[k];      a0[2 * j + 1] = (_Float16)s[k + 1];
            a1[2 * j] = (_Float16)s[32 + k]; a1[2 * j + 1] = (_Float16)s[33 + k];
        }
    }
#pragma unroll
    for (int c = 0; c < 2; ++c) {
        v16h b0 = loadBpk(pkWos, 2, 0, c);
        v16h b1 = loadBpk(pkWos, 2, 1, c);
        v8f acc = {};
        acc = wmma16(a0, b0, acc);
        acc = wmma16(a1, b1, acc);
#pragma unroll
        for (int i = 0; i < 8; ++i) {
            int r = n0 + i + 8 * hi;
            if (r < N) out[(size_t)r * 128 + c * 16 + nl] = acc[i] * sc;
        }
    }
#pragma unroll
    for (int d = 0; d < 3; ++d) {
        const float* vb = out_v + (size_t)rR * 192 + d;
#pragma unroll
        for (int j = 0; j < 8; ++j) {
            int k = akidx(j, hi);
            a0[2 * j] = (_Float16)vb[(size_t)k * 3];
            a0[2 * j + 1] = (_Float16)vb[(size_t)(k + 1) * 3];
            a1[2 * j] = (_Float16)vb[(size_t)(32 + k) * 3];
            a1[2 * j + 1] = (_Float16)vb[(size_t)(33 + k) * 3];
        }
#pragma unroll
        for (int c = 0; c < 2; ++c) {
            v16h b0 = loadBpk(pkWov, 2, 0, c);
            v16h b1 = loadBpk(pkWov, 2, 1, c);
            v8f acc = {};
            acc = wmma16(a0, b0, acc);
            acc = wmma16(a1, b1, acc);
#pragma unroll
            for (int i = 0; i < 8; ++i) {
                int r = n0 + i + 8 * hi;
                int w = c * 16 + nl;
                if (r < N) out[(size_t)r * 128 + 32 + w * 3 + d] = acc[i] * sc;
            }
        }
    }
}

// ---------------------------------------------------------------------------
extern "C" void kernel_launch(void* const* d_in, const int* in_sizes, int n_in,
                              void* d_out, int out_size, void* d_ws, size_t ws_size,
                              hipStream_t stream)
{
    const float* node_s    = (const float*)d_in[0];
    const float* node_v    = (const float*)d_in[1];
    const float* edge_attr = (const float*)d_in[2];
    const float* esa       = (const float*)d_in[3];
    const float* cutoff    = (const float*)d_in[4];
    const float* Wk0       = (const float*)d_in[5];
    const float* Wk1       = (const float*)d_in[6];
    const float* Wkp       = (const float*)d_in[7];
    const float* Wv0       = (const float*)d_in[8];
    const float* Wv1       = (const float*)d_in[9];
    const float* Wvp       = (const float*)d_in[10];
    const float* Wls       = (const float*)d_in[11];
    const float* Wlv       = (const float*)d_in[12];
    const float* Wos       = (const float*)d_in[13];
    const float* Wov       = (const float*)d_in[14];
    const int*   edge_src  = (const int*)d_in[15];
    const int*   edge_dst  = (const int*)d_in[16];

    const int N = in_sizes[0] / 32;   // MUL = 32
    const int E = in_sizes[16];

    // ---- workspace layout ----
    // packed f16 weights first (40960 halfs = 80KB), then float region
    _Float16* pk    = (_Float16*)d_ws;
    _Float16* pkWk0 = pk;             // 2048
    _Float16* pkWk1 = pk + 2048;      // 4096
    _Float16* pkWkp = pk + 6144;      // 8192
    _Float16* pkWv0 = pk + 14336;     // 2048
    _Float16* pkWv1 = pk + 16384;     // 4096
    _Float16* pkWvp = pk + 20480;     // 8192
    _Float16* pkWls = pk + 28672;     // 4096
    _Float16* pkWlv = pk + 32768;     // 4096
    _Float16* pkWos = pk + 36864;     // 2048
    _Float16* pkWov = pk + 38912;     // 2048
    float* fbase = (float*)(pk + 40960);
    // [lmax 2N][z 2N][out_s 64N][out_v 192N][logit 2E][ex 2E][wp_v 128E]
    float* lmax  = fbase;
    float* z     = lmax  + (size_t)2 * N;
    float* out_s = z     + (size_t)2 * N;
    float* out_v = out_s + (size_t)64 * N;
    float* logit = out_v + (size_t)192 * N;
    float* ex    = logit + (size_t)2 * E;
    float* wpv   = ex    + (size_t)2 * E;

    // ---- weight prepack (f32 -> WMMA-B f16 layout) ----
    kp_pack<<<(2048 + 255) / 256, 256, 0, stream>>>(Wk0, pkWk0, 32, 64);
    kp_pack<<<(4096 + 255) / 256, 256, 0, stream>>>(Wk1, pkWk1, 64, 64);
    kp_pack_path<<<(8192 + 255) / 256, 256, 0, stream>>>(Wkp, pkWkp);
    kp_pack<<<(2048 + 255) / 256, 256, 0, stream>>>(Wv0, pkWv0, 32, 64);
    kp_pack<<<(4096 + 255) / 256, 256, 0, stream>>>(Wv1, pkWv1, 64, 64);
    kp_pack_path<<<(8192 + 255) / 256, 256, 0, stream>>>(Wvp, pkWvp);
    kp_pack<<<(4096 + 255) / 256, 256, 0, stream>>>(Wls, pkWls, 32, 128);
    kp_pack<<<(4096 + 255) / 256, 256, 0, stream>>>(Wlv, pkWlv, 32, 128);
    kp_pack<<<(2048 + 255) / 256, 256, 0, stream>>>(Wos, pkWos, 64, 32);
    kp_pack<<<(2048 + 255) / 256, 256, 0, stream>>>(Wov, pkWov, 64, 32);

    // K0: zero the accumulators (first 260N floats of float region)
    size_t nzero = (size_t)260 * N;
    k0_zero<<<(unsigned)((nzero + 255) / 256), 256, 0, stream>>>(fbase, nzero);

    // K1: edge MLPs + TP(k) + logits (WMMA)
    int tiles = (E + 15) / 16;
    k1_edge_wmma<<<tiles, 32, 0, stream>>>(node_s, node_v, edge_attr, esa,
                                           pkWk0, pkWk1, pkWkp, pkWv0, pkWv1, pkWvp,
                                           pkWls, pkWlv, edge_src, edge_dst,
                                           wpv, logit, (unsigned int*)lmax, E);

    // K2: softmax numerators + denominators
    k2_softmax_den<<<(E * 2 + 255) / 256, 256, 0, stream>>>(
        logit, (const unsigned int*)lmax, cutoff, edge_dst, ex, z, E);

    // K3: weighted value scatter
    long long t3 = (long long)E * 64;
    k3_scatter<<<(unsigned)((t3 + 255) / 256), 256, 0, stream>>>(
        wpv, ex, z, node_s, node_v, edge_attr, edge_src, edge_dst, out_s, out_v, E);

    // K4: node output projection (WMMA)
    k4_node_wmma<<<(N + 15) / 16, 32, 0, stream>>>(out_s, out_v, pkWos, pkWov,
                                                   (float*)d_out, N);
}
